// Quantizer_70214125355581
// MI455X (gfx1250) — compile-verified, matrix operand order checked by reading, not run on hardware
//
#include <hip/hip_runtime.h>
#include <hip/hip_bf16.h>

#define NROWS 65536
#define DDIM  512
#define KCODE 1024
#define NT    (KCODE / 16)

typedef __attribute__((ext_vector_type(16))) __bf16 v16bf;
typedef __attribute__((ext_vector_type(8)))  float  v8f;
typedef __attribute__((ext_vector_type(4)))  int    v4i;

#if defined(__has_builtin)
# if __has_builtin(__builtin_amdgcn_global_load_async_to_lds_b128)
#  define HAVE_ASYNC_COPY 1
# endif
# if __has_builtin(__builtin_amdgcn_s_wait_asynccnt)
#  define HAVE_WAIT_ASYNC 1
# endif
#endif

#if defined(HAVE_ASYNC_COPY)
typedef __attribute__((address_space(1))) v4i gv4i_t;   // global (AS1)
typedef __attribute__((address_space(3))) v4i lv4i_t;   // LDS (AS3)
__device__ __forceinline__ void async_copy_b128(const void* g, void* l) {
    __builtin_amdgcn_global_load_async_to_lds_b128(
        (gv4i_t*)g, (lv4i_t*)l, 0, 0);
}
#endif

template <int N>
__device__ __forceinline__ void wait_async() {
#if defined(HAVE_ASYNC_COPY)
# if defined(HAVE_WAIT_ASYNC)
    __builtin_amdgcn_s_wait_asynccnt(N);
# else
    if (N == 4) asm volatile("s_wait_asynccnt 0x4" ::: "memory");
    else        asm volatile("s_wait_asynccnt 0x0" ::: "memory");
# endif
#endif
}

__device__ __forceinline__ unsigned bf16rne(float f) {
    union { float f; unsigned u; } c; c.f = f;
    unsigned u = c.u;
    u += 0x7FFFu + ((u >> 16) & 1u);   // round-to-nearest-even
    return u >> 16;
}
__device__ __forceinline__ unsigned pack2bf(float lo, float hi) {
    return bf16rne(lo) | (bf16rne(hi) << 16);
}

// ---------------------------------------------------------------------------
// Kernel 1: wnorm[k] = sum(weight[k]^2); weight f32 -> bf16 (workspace)
// ---------------------------------------------------------------------------
__global__ __launch_bounds__(256) void vq_prep(const float* __restrict__ w,
                                               unsigned short* __restrict__ wbf,
                                               float* __restrict__ wnorm) {
    const int wave = threadIdx.x >> 5;
    const int lane = threadIdx.x & 31;
    const int row  = blockIdx.x * 8 + wave;
    const float4* src = (const float4*)(w + (size_t)row * DDIM);
    unsigned* dst = (unsigned*)(wbf + (size_t)row * DDIM);
    float ss = 0.0f;
#pragma unroll
    for (int i = 0; i < 4; ++i) {
        int j = i * 32 + lane;
        float4 v = src[j];
        ss += v.x * v.x + v.y * v.y + v.z * v.z + v.w * v.w;
        dst[j * 2 + 0] = pack2bf(v.x, v.y);
        dst[j * 2 + 1] = pack2bf(v.z, v.w);
    }
#pragma unroll
    for (int off = 16; off >= 1; off >>= 1) ss += __shfl_xor(ss, off, 32);
    if (lane == 0) wnorm[row] = ss;
}

// ---------------------------------------------------------------------------
// Kernel 2: score = wnorm[k] - 2*h.w_k via bf16 WMMA; argmin over k.
// Block = 8 waves, covers 256 rows: each wave holds TWO 16x512 A panels in
// VGPRs (rows base, base+128) so every B fragment from LDS feeds 2 WMMAs.
// B tiles double-buffered in LDS, staged with GLOBAL_LOAD_ASYNC_TO_LDS.
// ---------------------------------------------------------------------------
__global__ __launch_bounds__(256) void vq_argmin(const float* __restrict__ h,
                                                 const unsigned short* __restrict__ wbf,
                                                 const float* __restrict__ wnorm,
                                                 int* __restrict__ closest,
                                                 float* __restrict__ idx_out) {
    __shared__ unsigned short Bs[2][16][DDIM + 8];
    __shared__ float WN[KCODE];

    const int tid  = threadIdx.x;
    const int wave = tid >> 5;
    const int lane = tid & 31;
    const int g    = lane >> 4;
    const int l    = lane & 15;

    for (int i = tid; i < KCODE; i += 256) WN[i] = wnorm[i];

    // cooperative staging role: 64B per thread
    const int sr = tid >> 4;             // tile row 0..15
    const int sc = (tid & 15) * 32;      // 32 bf16 per thread

    // ---- prologue: stage tile 0 into buffer 0 ----
    {
        const uint4* src = (const uint4*)(wbf + (size_t)sr * DDIM + sc);
#if defined(HAVE_ASYNC_COPY)
#pragma unroll
        for (int i = 0; i < 4; ++i) async_copy_b128(src + i, &Bs[0][sr][sc + i * 8]);
#else
        uint4* dst = (uint4*)&Bs[0][sr][sc];
#pragma unroll
        for (int i = 0; i < 4; ++i) dst[i] = src[i];
#endif
    }

    // ---- load both A panels (rows rowBase+l and rowBase+128+l) ----
    const int rowBase = blockIdx.x * 256 + wave * 16;
    const float* hrow0 = h + (size_t)(rowBase + l) * DDIM;
    const float* hrow1 = h + (size_t)(rowBase + 128 + l) * DDIM;
    v16bf A0[16], A1[16];
#pragma unroll
    for (int s = 0; s < 16; ++s) {
        const int d0 = 32 * s + g * 8;          // ISA 16-bit A layout
        union { v16bf v; unsigned u[8]; } af;
        {
            const float4* p0 = (const float4*)(hrow0 + d0);
            const float4* p1 = (const float4*)(hrow0 + d0 + 16);
            float4 a0 = p0[0], a1 = p0[1], b0 = p1[0], b1 = p1[1];
            af.u[0] = pack2bf(a0.x, a0.y);  af.u[1] = pack2bf(a0.z, a0.w);
            af.u[2] = pack2bf(a1.x, a1.y);  af.u[3] = pack2bf(a1.z, a1.w);
            af.u[4] = pack2bf(b0.x, b0.y);  af.u[5] = pack2bf(b0.z, b0.w);
            af.u[6] = pack2bf(b1.x, b1.y);  af.u[7] = pack2bf(b1.z, b1.w);
            A0[s] = af.v;
        }
        {
            const float4* p0 = (const float4*)(hrow1 + d0);
            const float4* p1 = (const float4*)(hrow1 + d0 + 16);
            float4 a0 = p0[0], a1 = p0[1], b0 = p1[0], b1 = p1[1];
            af.u[0] = pack2bf(a0.x, a0.y);  af.u[1] = pack2bf(a0.z, a0.w);
            af.u[2] = pack2bf(a1.x, a1.y);  af.u[3] = pack2bf(a1.z, a1.w);
            af.u[4] = pack2bf(b0.x, b0.y);  af.u[5] = pack2bf(b0.z, b0.w);
            af.u[6] = pack2bf(b1.x, b1.y);  af.u[7] = pack2bf(b1.z, b1.w);
            A1[s] = af.v;
        }
    }

    float runmin0[8], runmin1[8];
    int   runidx0[8], runidx1[8];
#pragma unroll
    for (int v = 0; v < 8; ++v) {
        runmin0[v] = 3.4e38f; runidx0[v] = 0;
        runmin1[v] = 3.4e38f; runidx1[v] = 0;
    }

    for (int kt = 0; kt < NT; ++kt) {
        const int buf = kt & 1;
        // ---- stage tile kt+1 into the other buffer (async) ----
        if (kt + 1 < NT) {
            const uint4* src = (const uint4*)(wbf + (size_t)((kt + 1) * 16 + sr) * DDIM + sc);
#if defined(HAVE_ASYNC_COPY)
#pragma unroll
            for (int i = 0; i < 4; ++i)
                async_copy_b128(src + i, &Bs[buf ^ 1][sr][sc + i * 8]);
#else
            uint4* dst = (uint4*)&Bs[buf ^ 1][sr][sc];
#pragma unroll
            for (int i = 0; i < 4; ++i) dst[i] = src[i];
#endif
            wait_async<4>();               // tile kt's copies (this wave) complete
        } else {
            wait_async<0>();
        }
        __syncthreads();                   // tile kt visible to all waves

        v8f acc0 = {}, acc1 = {};
#pragma unroll
        for (int s = 0; s < 16; ++s) {
            const int d0 = 32 * s + g * 8;
            union { v16bf v; uint4 q[2]; } bf;
            bf.q[0] = *(const uint4*)&Bs[buf][l][d0];
            bf.q[1] = *(const uint4*)&Bs[buf][l][d0 + 16];
            acc0 = __builtin_amdgcn_wmma_f32_16x16x32_bf16(
                false, A0[s], false, bf.v, (short)0, acc0, false, false);
            acc1 = __builtin_amdgcn_wmma_f32_16x16x32_bf16(
                false, A1[s], false, bf.v, (short)0, acc1, false, false);
        }

        const int   col = kt * 16 + l;
        const float wn  = WN[col];
#pragma unroll
        for (int v = 0; v < 8; ++v) {
            float s0 = wn - 2.0f * acc0[v];
            if (s0 < runmin0[v]) { runmin0[v] = s0; runidx0[v] = col; }
            float s1 = wn - 2.0f * acc1[v];
            if (s1 < runmin1[v]) { runmin1[v] = s1; runidx1[v] = col; }
        }
        __syncthreads();                   // all waves done with buf before reuse
    }

    // ---- min+argmin across the 16 lanes holding one row; write results ----
#pragma unroll
    for (int p = 0; p < 2; ++p) {
#pragma unroll
        for (int v = 0; v < 8; ++v) {
            float m  = p ? runmin1[v] : runmin0[v];
            int   ix = p ? runidx1[v] : runidx0[v];
#pragma unroll
            for (int off = 1; off < 16; off <<= 1) {
                float om = __shfl_xor(m, off, 32);
                int   oi = __shfl_xor(ix, off, 32);
                if (om < m || (om == m && oi < ix)) { m = om; ix = oi; }
            }
            if (l == 0) {
                const int row = rowBase + p * 128 + g * 8 + v;
                closest[row] = ix;
                idx_out[row] = (float)ix;
            }
        }
    }
}

// ---------------------------------------------------------------------------
// Kernel 3: gather codeword (f32), write quantized_st, per-row loss.
// ---------------------------------------------------------------------------
__global__ __launch_bounds__(256) void vq_finalize(const float* __restrict__ h,
                                                   const float* __restrict__ w,
                                                   const int* __restrict__ closest,
                                                   float* __restrict__ qout,
                                                   float* __restrict__ loss) {
    const int wave = threadIdx.x >> 5;
    const int lane = threadIdx.x & 31;
    const int row  = blockIdx.x * 8 + wave;
    const int idx  = closest[row];
    const float4* wr = (const float4*)(w + (size_t)idx * DDIM);
    const float4* hr = (const float4*)(h + (size_t)row * DDIM);
    float4*       qr = (float4*)(qout + (size_t)row * DDIM);
    float ss = 0.0f;
#pragma unroll
    for (int i = 0; i < 4; ++i) {
        int j = i * 32 + lane;
        float4 wv = wr[j];
        float4 hv = hr[j];
        qr[j] = wv;
        float dx = wv.x - hv.x, dy = wv.y - hv.y, dz = wv.z - hv.z, dw = wv.w - hv.w;
        ss += dx * dx + dy * dy + dz * dz + dw * dw;
    }
#pragma unroll
    for (int off = 16; off >= 1; off >>= 1) ss += __shfl_xor(ss, off, 32);
    if (lane == 0)
        loss[row] = (0.1f + 0.2f) * ss * (1.0f / (float)DDIM);
}

// ---------------------------------------------------------------------------
extern "C" void kernel_launch(void* const* d_in, const int* in_sizes, int n_in,
                              void* d_out, int out_size, void* d_ws, size_t ws_size,
                              hipStream_t stream) {
    const float* h = (const float*)d_in[0];         // (N, D) f32
    const float* w = (const float*)d_in[1];         // (K, D) f32

    float* q_out    = (float*)d_out;                // N*D  quantized_st
    float* idx_out  = q_out + (size_t)NROWS * DDIM; // N    indices (as float)
    float* loss_out = idx_out + NROWS;              // N    loss

    char* ws = (char*)d_ws;
    float*          wnorm   = (float*)ws;                                   // 4 KB
    unsigned short* wbf     = (unsigned short*)(ws + 4096);                 // 1 MB
    int*            closest = (int*)(ws + 4096 + (size_t)KCODE * DDIM * 2); // 256 KB

    vq_prep    <<<KCODE / 8,   256, 0, stream>>>(w, wbf, wnorm);
    vq_argmin  <<<NROWS / 256, 256, 0, stream>>>(h, wbf, wnorm, closest, idx_out);
    vq_finalize<<<NROWS / 8,   256, 0, stream>>>(h, w, closest, q_out, loss_out);
}
